// PhaseFormerBlock_85521388798328
// MI455X (gfx1250) — matrix operand order repaired and checked by reading, not compiled, not run on hardware
//
#include <hip/hip_runtime.h>
#include <math.h>

#define DEV __device__ __forceinline__

typedef _Float16 half_t;
typedef __attribute__((ext_vector_type(16))) _Float16 v16h;
typedef __attribute__((ext_vector_type(8)))  float    v8f;

// Problem constants (match reference)
constexpr int NB = 64, NC = 32, NL = 96, NPIN = 128, NPOUT = 128;
constexpr int ND = 128, NA = 64, NH = 8, NR = 8, NDH = 8;
constexpr float SCALE = 0.35355339059327373f;  // 1/sqrt(8)
constexpr size_t Z_ELEMS = (size_t)NB * NC * NL * ND;  // 25165824

// ---- dynamic LDS carve (bytes) ----
constexpr int OFF_PH  = 0;        // phase f16 [96][128] = 24576 B ; later aliased as z_new f16
constexpr int OFF_Z16 = 24576;    // z f16   [96][128] = 24576 B
constexpr int OFF_Z32 = 49152;    // z f32   [96][128] = 49152 B (residual)
constexpr int OFF_K   = 98304;    // k  f16  [96][64]  = 12288 B
constexpr int OFF_V   = 110592;   // v  f16  [96][64]
constexpr int OFF_Q2  = 122880;   // q2 f16  [96][64]
constexpr int OFF_AO  = 135168;   // attn-out f16 [96][64]
constexpr int OFF_RST = 147456;   // rstate f32 [8][64] = 2048 B
constexpr int OFF_K2  = 149504;   // k2 f32 [8][64]
constexpr int OFF_V2  = 151552;   // v2 f32 [8][64]
constexpr int SMEM_BYTES = 153600;

// A-fragment (16x32 f16, M x K) from row-major f16 LDS matrix, leading dim ld.
// ISA layout: lane m = lane&15 ; kb = (lane>>4)*8 ; elems 0..7 -> K kb..kb+7,
// elems 8..15 -> K kb+16..kb+23.
DEV v16h load_a_frag(const half_t* S, int ld, int m0, int k0, int lane) {
  const int m  = m0 + (lane & 15);
  const int kb = k0 + ((lane >> 4) << 3);
  const half_t* p = S + m * ld + kb;
  v16h a;
#pragma unroll
  for (int e = 0; e < 8; ++e) { a[e] = p[e]; a[e + 8] = p[e + 16]; }
  return a;
}

// B-fragment (32x16 f16, K x N) gathered from row-major f32 global weight,
// leading dim ld. ISA layout: n = lane&15 ; kb = (lane>>4)*8 ; elems 0..7 -> K kb..kb+7,
// elems 8..15 -> K kb+16..kb+23.
DEV v16h load_b_frag(const float* W, int ld, int k0, int n0, int lane) {
  const int n  = n0 + (lane & 15);
  const int kb = k0 + ((lane >> 4) << 3);
  const float* p = W + (size_t)kb * ld + n;
  v16h b;
#pragma unroll
  for (int e = 0; e < 8; ++e) { b[e] = (half_t)p[(size_t)e * ld]; b[e + 8] = (half_t)p[(size_t)(e + 16) * ld]; }
  return b;
}

DEV v8f wmma16(v16h a, v16h b, v8f c) {
  return __builtin_amdgcn_wmma_f32_16x16x32_f16(false, a, false, b, (short)0, c,
                                                false, false);
}

DEV float wred_sum(float v) {
#pragma unroll
  for (int off = 16; off > 0; off >>= 1) v += __shfl_xor(v, off, 32);
  return v;
}
DEV float wred_max(float v) {
#pragma unroll
  for (int off = 16; off > 0; off >>= 1) v = fmaxf(v, __shfl_xor(v, off, 32));
  return v;
}

DEV int center_of(int r) { return (int)rintf((float)r * (95.0f / 7.0f)); }

__global__ void __launch_bounds__(256)
phaseformer_kernel(const float* __restrict__ phase,
                   const float* __restrict__ enc_W, const float* __restrict__ enc_b,
                   const float* __restrict__ Wk,    const float* __restrict__ Wv,
                   const float* __restrict__ routers,
                   const float* __restrict__ Wq2,   const float* __restrict__ Wk2,
                   const float* __restrict__ Wv2,   const float* __restrict__ Wo,
                   const float* __restrict__ relpos,
                   const float* __restrict__ ln_g,  const float* __restrict__ ln_b,
                   const float* __restrict__ dec_W, const float* __restrict__ dec_b,
                   float* __restrict__ out_z, float* __restrict__ out_y) {
  extern __shared__ __align__(16) char smem[];
  half_t* sPh  = (half_t*)(smem + OFF_PH);
  half_t* sZ16 = (half_t*)(smem + OFF_Z16);
  float*  sZ32 = (float*)(smem + OFF_Z32);
  half_t* sK   = (half_t*)(smem + OFF_K);
  half_t* sV   = (half_t*)(smem + OFF_V);
  half_t* sQ2  = (half_t*)(smem + OFF_Q2);
  half_t* sAO  = (half_t*)(smem + OFF_AO);
  float*  sRst = (float*)(smem + OFF_RST);
  float*  sK2  = (float*)(smem + OFF_K2);
  float*  sV2  = (float*)(smem + OFF_V2);

  const int bc   = blockIdx.x;          // 0..2047
  const int tid  = threadIdx.x;
  const int lane = tid & 31;
  const int wave = tid >> 5;

  // ---- 1) stage phase tile [96][128] f32 -> f16 LDS (coalesced) ----
  const float* Pblk = phase + (size_t)bc * NL * NPIN;
  for (int i = tid; i < NL * NPIN; i += 256) sPh[i] = (half_t)Pblk[i];
  __syncthreads();

  // ---- 2) encoder: z = Ph @ enc_W + enc_b  ([96,128]@[128,128]) ----
  {
    const int n0 = wave * 16;
    v16h bf[4];
#pragma unroll
    for (int ks = 0; ks < 4; ++ks) bf[ks] = load_b_frag(enc_W, ND, ks * 32, n0, lane);
    const int   d  = n0 + (lane & 15);
    const float eb = enc_b[d];
    for (int mt = 0; mt < 6; ++mt) {
      v8f acc = {};
#pragma unroll
      for (int ks = 0; ks < 4; ++ks)
        acc = wmma16(load_a_frag(sPh, NPIN, mt * 16, ks * 32, lane), bf[ks], acc);
#pragma unroll
      for (int i = 0; i < 8; ++i) {
        const int m = mt * 16 + ((lane >> 4) << 3) + i;
        const float zv = acc[i] + eb;
        sZ32[m * ND + d] = zv;
        sZ16[m * ND + d] = (half_t)zv;
      }
    }
  }
  __syncthreads();

  // ---- 3) k / v / q2 projections ([96,128]@[128,64] each) ----
  for (int ct = wave; ct < 12; ct += 8) {
    const float* W = (ct < 4) ? Wk : (ct < 8) ? Wv : Wq2;
    half_t* Od     = (ct < 4) ? sK : (ct < 8) ? sV : sQ2;
    const int n0 = (ct & 3) * 16;
    v16h bf[4];
#pragma unroll
    for (int ks = 0; ks < 4; ++ks) bf[ks] = load_b_frag(W, NA, ks * 32, n0, lane);
    const int n = n0 + (lane & 15);
    for (int mt = 0; mt < 6; ++mt) {
      v8f acc = {};
#pragma unroll
      for (int ks = 0; ks < 4; ++ks)
        acc = wmma16(load_a_frag(sZ16, ND, mt * 16, ks * 32, lane), bf[ks], acc);
#pragma unroll
      for (int i = 0; i < 8; ++i) {
        const int m = mt * 16 + ((lane >> 4) << 3) + i;
        Od[m * NA + n] = (half_t)acc[i];
      }
    }
  }
  __syncthreads();

  // ---- 4) router attention: softmax over L, rstate[r][h*8+dh] ----
  for (int iter = 0; iter < 8; ++iter) {
    const int pr = iter * 8 + wave;        // (h,r) pair 0..63
    const int h = pr >> 3, r = pr & 7;
    const int cr = center_of(r);
    float rq[8];
#pragma unroll
    for (int d = 0; d < 8; ++d) rq[d] = routers[r * NA + h * 8 + d];
    float sv[3], mx = -3.0e38f;
#pragma unroll
    for (int j = 0; j < 3; ++j) {
      const int l = lane + 32 * j;
      float s = 0.f;
#pragma unroll
      for (int d = 0; d < 8; ++d) s += rq[d] * (float)sK[l * NA + h * 8 + d];
      s = s * SCALE + relpos[h * 191 + (l - cr + 95)];
      sv[j] = s; mx = fmaxf(mx, s);
    }
    mx = wred_max(mx);
    float sum = 0.f;
#pragma unroll
    for (int j = 0; j < 3; ++j) { sv[j] = __expf(sv[j] - mx); sum += sv[j]; }
    sum = wred_sum(sum);
    const float inv = 1.f / sum;
#pragma unroll
    for (int d = 0; d < 8; ++d) {
      float p = 0.f;
#pragma unroll
      for (int j = 0; j < 3; ++j) {
        const int l = lane + 32 * j;
        p += sv[j] * (float)sV[l * NA + h * 8 + d];
      }
      p = wred_sum(p);
      if (lane == 0) sRst[r * NA + h * 8 + d] = p * inv;
    }
  }
  __syncthreads();

  // ---- 5) k2 = rstate@Wk2, v2 = rstate@Wv2  ([8,64]@[64,64]) ----
  for (int idx = tid; idx < NR * NA; idx += 256) {
    const int r = idx >> 6, e = idx & 63;
    float aK = 0.f, aV = 0.f;
    for (int a = 0; a < NA; ++a) {
      const float rv = sRst[r * NA + a];
      aK += rv * Wk2[a * NA + e];
      aV += rv * Wv2[a * NA + e];
    }
    sK2[idx] = aK; sV2[idx] = aV;
  }
  __syncthreads();

  // ---- 6) second attention per (h,l): softmax over R=8 ----
  for (int idx = tid; idx < NH * NL; idx += 256) {
    const int h = idx & 7, l = idx >> 3;
    float q[8];
#pragma unroll
    for (int d = 0; d < 8; ++d) q[d] = (float)sQ2[l * NA + h * 8 + d];
    float s[8], mx = -3.0e38f;
#pragma unroll
    for (int r = 0; r < 8; ++r) {
      float t = 0.f;
#pragma unroll
      for (int d = 0; d < 8; ++d) t += q[d] * sK2[r * NA + h * 8 + d];
      t = t * SCALE + relpos[h * 191 + (l - center_of(r) + 95)];
      s[r] = t; mx = fmaxf(mx, t);
    }
    float sum = 0.f;
#pragma unroll
    for (int r = 0; r < 8; ++r) { s[r] = __expf(s[r] - mx); sum += s[r]; }
    const float inv = 1.f / sum;
#pragma unroll
    for (int d = 0; d < 8; ++d) {
      float o = 0.f;
#pragma unroll
      for (int r = 0; r < 8; ++r) o += s[r] * sV2[r * NA + h * 8 + d];
      sAO[l * NA + h * 8 + d] = (half_t)(o * inv);
    }
  }
  __syncthreads();

  // ---- 7) out-proj + residual: h = z + AO@Wo  ([96,64]@[64,128]) ----
  {
    const int n0 = wave * 16;
    v16h bf[2];
#pragma unroll
    for (int ks = 0; ks < 2; ++ks) bf[ks] = load_b_frag(Wo, ND, ks * 32, n0, lane);
    const int d = n0 + (lane & 15);
    for (int mt = 0; mt < 6; ++mt) {
      v8f acc = {};
#pragma unroll
      for (int ks = 0; ks < 2; ++ks)
        acc = wmma16(load_a_frag(sAO, NA, mt * 16, ks * 32, lane), bf[ks], acc);
#pragma unroll
      for (int i = 0; i < 8; ++i) {
        const int m = mt * 16 + ((lane >> 4) << 3) + i;
        sZ32[m * ND + d] += acc[i];
      }
    }
  }
  __syncthreads();

  // ---- 8) LayerNorm -> out_z (f32) + z_new f16 (alias over sPh) ----
  half_t* sZn = sPh;
  if (tid < NL) {
    const int l = tid;
    float mu = 0.f;
    for (int d = 0; d < ND; ++d) mu += sZ32[l * ND + d];
    mu *= (1.f / ND);
    float var = 0.f;
    for (int d = 0; d < ND; ++d) { const float t = sZ32[l * ND + d] - mu; var += t * t; }
    var *= (1.f / ND);
    const float rs = rsqrtf(var + 1e-5f);
    const size_t rowbase = ((size_t)bc * NL + l) * ND;
    for (int d = 0; d < ND; ++d) {
      const float zn = (sZ32[l * ND + d] - mu) * rs * ln_g[d] + ln_b[d];
      out_z[rowbase + d] = zn;
      sZn[l * ND + d] = (half_t)zn;
    }
  }
  __syncthreads();

  // ---- 9) decoder: y = z_new @ dec_W + dec_b  ([96,128]@[128,128]) ----
  {
    const int n0 = wave * 16;
    v16h bf[4];
#pragma unroll
    for (int ks = 0; ks < 4; ++ks) bf[ks] = load_b_frag(dec_W, NPOUT, ks * 32, n0, lane);
    const int n = n0 + (lane & 15);
    const float db = dec_b[n];
    for (int mt = 0; mt < 6; ++mt) {
      v8f acc = {};
#pragma unroll
      for (int ks = 0; ks < 4; ++ks)
        acc = wmma16(load_a_frag(sZn, ND, mt * 16, ks * 32, lane), bf[ks], acc);
#pragma unroll
      for (int i = 0; i < 8; ++i) {
        const int m = mt * 16 + ((lane >> 4) << 3) + i;
        out_y[((size_t)bc * NL + m) * NPOUT + n] = acc[i] + db;
      }
    }
  }
}

extern "C" void kernel_launch(void* const* d_in, const int* in_sizes, int n_in,
                              void* d_out, int out_size, void* d_ws, size_t ws_size,
                              hipStream_t stream) {
  (void)in_sizes; (void)n_in; (void)out_size; (void)d_ws; (void)ws_size;

  const float* phase   = (const float*)d_in[0];
  const float* enc_W   = (const float*)d_in[1];
  const float* enc_b   = (const float*)d_in[2];
  const float* Wk      = (const float*)d_in[3];
  const float* Wv      = (const float*)d_in[4];
  const float* routers = (const float*)d_in[5];
  const float* Wq2     = (const float*)d_in[6];
  const float* Wk2     = (const float*)d_in[7];
  const float* Wv2     = (const float*)d_in[8];
  const float* Wo      = (const float*)d_in[9];
  const float* relpos  = (const float*)d_in[10];
  const float* ln_g    = (const float*)d_in[11];
  const float* ln_b    = (const float*)d_in[12];
  const float* dec_W   = (const float*)d_in[13];
  const float* dec_b   = (const float*)d_in[14];

  float* out_z = (float*)d_out;
  float* out_y = (float*)d_out + Z_ELEMS;

  // Allow >48KB dynamic LDS (no-op if already permitted).
  (void)hipFuncSetAttribute((const void*)phaseformer_kernel,
                            hipFuncAttributeMaxDynamicSharedMemorySize, SMEM_BYTES);

  phaseformer_kernel<<<dim3(NB * NC), dim3(256), SMEM_BYTES, stream>>>(
      phase, enc_W, enc_b, Wk, Wv, routers, Wq2, Wk2, Wv2, Wo, relpos,
      ln_g, ln_b, dec_W, dec_b, out_z, out_y);
}
